// ToepliztV4_76106820485420
// MI455X (gfx1250) — matrix operand order repaired and checked by reading, not compiled, not run on hardware
//
#include <hip/hip_runtime.h>
#include <hip/hip_bf16.h>
#include <stdint.h>

#define H     8
#define SEQN  4096
#define DIM   64
#define BATCH 8
#define NCH   (H * DIM)              // 512 channels
#define APAD  (64 + SEQN + 64)       // 4224: guarded (reversed) kernel row, bf16
#define SPAD  (64 + SEQN)            // 4160: guarded x row, bf16

#define A_BYTES ((size_t)NCH * APAD * 2)                 // 4,325,376 (16B aligned)
#define X_OFF   A_BYTES
#define X_BYTES ((size_t)NCH * BATCH * SPAD * 2)         // 34,078,720
#define WS_USED (X_OFF + X_BYTES)

typedef __attribute__((ext_vector_type(16))) __bf16 v16bf;
typedef __attribute__((ext_vector_type(8)))  float  v8f;

static __device__ __forceinline__ unsigned short f32_to_bf16(float f) {
    union { float f; unsigned int u; } c; c.f = f;
    unsigned int r = c.u + 0x7fffu + ((c.u >> 16) & 1u);   // round-to-nearest-even
    return (unsigned short)(r >> 16);
}

// ---------------------------------------------------------------------------
// Kernel 1: REVERSED kernel row.  arev[ch][4096 - l] = exp(clamp(...)), so the
// Toeplitz A-tile gather reads ASCENDING contiguous runs (two ds_load_b128,
// no byte-reversal movs).  Guards ([0] and [4097..]) are zero from the memset
// -> causal masking for free.
// ---------------------------------------------------------------------------
__global__ void prep_a(const float* __restrict__ zero_, const float* __restrict__ pos,
                       const float* __restrict__ gamma, unsigned short* __restrict__ abf) {
    int ch = blockIdx.x;                 // h*DIM + d
    int d  = ch & (DIM - 1);
    int h  = ch >> 6;
    float lg = __logf(gamma[ch]);
    unsigned short* row = abf + (size_t)ch * APAD;
    for (int l = threadIdx.x; l < SEQN; l += blockDim.x) {
        float v = (l == 0) ? zero_[ch]
                           : fmaf(lg, (float)l, pos[((size_t)h * (SEQN - 1) + (l - 1)) * DIM + d]);
        v = fminf(fmaxf(v, -60.f), 30.f);
        row[SEQN - l] = f32_to_bf16(__expf(v));   // reversed storage
    }
}

// ---------------------------------------------------------------------------
// Kernel 2: transpose x [B,H,N,DIM] f32 -> xbf [ch][b][SPAD] bf16 (contig in s)
// via LDS tile so global reads stay coalesced along d.
// ---------------------------------------------------------------------------
__global__ void prep_x(const float* __restrict__ x, unsigned short* __restrict__ xbf) {
    __shared__ unsigned short tile[32][DIM + 2];
    int bx = blockIdx.x;                  // h*B*128 + b*128 + sb
    int sb = bx & 127, b = (bx >> 7) & 7, h = bx >> 10;
    int s0 = sb << 5;
    int tid = threadIdx.x;                // 256 threads
    const float* xp = x + (((size_t)b * H + h) * SEQN + s0) * DIM;
    #pragma unroll
    for (int i = 0; i < 8; ++i) {
        int idx = tid + (i << 8);
        int si = idx >> 6, dd = idx & 63;
        tile[si][dd] = f32_to_bf16(xp[(size_t)si * DIM + dd]);
    }
    __syncthreads();
    int sp = tid & 31, dq = tid >> 5;
    #pragma unroll
    for (int p = 0; p < 8; ++p) {
        int dd = (p << 3) + dq;
        size_t off = ((size_t)(h * DIM + dd) * BATCH + b) * SPAD + 64 + s0 + sp;
        xbf[off] = tile[sp][dd];
    }
}

// ---------------------------------------------------------------------------
// Kernel 3: causal Toeplitz multiply, v_wmma_f32_16x16x32_bf16.
// Each wave owns 4 column-groups g = 4q..4q+3 (64 output columns).  All four
// share one Toeplitz A tile per lag window, and their B windows telescope:
// B(j,w) = T(4q+j-w), so per iteration we load ONE fresh B tile and rotate.
// Steady state (unroll 4): 8 ds_b128 + 8 global_b128 + 16 WMMA per block.
// q is assigned LARGEST-FIRST across the dispatch to kill the tail.
// ---------------------------------------------------------------------------
union BV { uint4 u2[2]; v16bf v; };

__global__ void __launch_bounds__(128) toeplitz_wmma(
        const unsigned short* __restrict__ abf,
        const unsigned short* __restrict__ xbf,
        float* __restrict__ out) {
    __shared__ __align__(16) unsigned short aLds[APAD];
    int ch = blockIdx.x;
    int d  = ch & 63, h = ch >> 6;

    {   // stage reversed kernel row (8448 B) cooperatively
        const uint4* src = (const uint4*)(abf + (size_t)ch * APAD);
        uint4* dst = (uint4*)aLds;
        for (int i = threadIdx.x; i < (APAD * 2 / 16); i += blockDim.x) dst[i] = src[i];
    }
    __syncthreads();

    int lane = threadIdx.x & 31;
    int wave = __builtin_amdgcn_readfirstlane(threadIdx.x >> 5);  // SGPR loop ctrl
    int q    = 31 - (blockIdx.y * 4 + wave);   // LPT: biggest waves launch first

    int cN  = lane & 15;                    // B column index == A row M
    int bC  = cN & 7;                       // batch of this column
    int hiK = (lane >> 4) << 4;             // K offset for B half (0/16)
    int klo = (lane >> 4) << 4;             // K base for A half  (0/16)
    int c16 = (cN >> 3) << 4;               // chunk select within group

    // tile T(m): 16 consecutive bf16 at xcol + 32*m  (s0 = c16 - 16 + 32m)
    const unsigned short* xcol =
        xbf + ((size_t)ch * BATCH + bC) * SPAD + 64 + hiK + c16 - 16;

    // Toeplitz A tile for window w (reversed layout -> ascending runs)
    auto loadA = [&](int w) {
        union { unsigned short u[16]; v16bf v; } A;
        const unsigned short* ap = aLds + (4080 - (w << 5) - cN + klo);
        #pragma unroll
        for (int e = 0; e < 8; ++e) { A.u[e] = ap[e]; A.u[e + 8] = ap[e + 8]; }
        return A.v;
    };
    auto loadB = [&](int m) {
        BV b; const uint4* p = (const uint4*)(xcol + (m << 5));
        b.u2[0] = p[0]; b.u2[1] = p[1]; return b;
    };
    auto mm = [&](v16bf A, const BV& B, v8f C) {
        return __builtin_amdgcn_wmma_f32_16x16x32_bf16(false, A, false, B.v,
                                                       (short)0, C, false, false);
    };

    v8f acc0 = {}, acc1 = {}, acc2 = {}, acc3 = {};
    int g0 = q << 2;                         // group 0 index (SGPR)
    BV r1 = loadB(g0 + 1), r2 = loadB(g0 + 2), r3 = loadB(g0 + 3);

    #pragma unroll 4
    for (int w = 0; w <= g0; ++w) {          // uniform trip count -> s_cmp loop
        BV r0 = loadB(g0 - w);               // one fresh B tile per iteration
        v16bf A = loadA(w);
        acc0 = mm(A, r0, acc0);
        acc1 = mm(A, r1, acc1);
        acc2 = mm(A, r2, acc2);
        acc3 = mm(A, r3, acc3);
        r3 = r2; r2 = r1; r1 = r0;           // telescoping rotation (renamed away)
    }
    {   v16bf A = loadA(g0 + 1);             // peeled: groups 1..3
        acc1 = mm(A, r1, acc1); acc2 = mm(A, r2, acc2); acc3 = mm(A, r3, acc3);
        r3 = r2; r2 = r1; }
    {   v16bf A = loadA(g0 + 2);             // groups 2..3
        acc2 = mm(A, r2, acc2); acc3 = mm(A, r3, acc3);
        r3 = r2; }
    {   v16bf A = loadA(g0 + 3);             // group 3
        acc3 = mm(A, r3, acc3); }

    // D layout: lane = column, VGPR r = row r (lanes 0-15) or r+8 (lanes 16-31)
    int rbase = (lane >> 4) << 3;
    size_t obase = ((size_t)bC * H + h) * SEQN;
    v8f accs[4] = {acc0, acc1, acc2, acc3};
    #pragma unroll
    for (int j = 0; j < 4; ++j) {
        int t0 = ((g0 + j) << 5) + c16;
        #pragma unroll
        for (int r = 0; r < 8; ++r)
            __builtin_nontemporal_store(accs[j][r],
                &out[(obase + t0 + rbase + r) * DIM + d]);
    }
}

// ---------------------------------------------------------------------------
extern "C" void kernel_launch(void* const* d_in, const int* in_sizes, int n_in,
                              void* d_out, int out_size, void* d_ws, size_t ws_size,
                              hipStream_t stream) {
    const float* x     = (const float*)d_in[0];
    const float* zero_ = (const float*)d_in[1];
    const float* pos   = (const float*)d_in[2];
    const float* gamma = (const float*)d_in[3];
    float* out = (float*)d_out;

    unsigned short* abf = (unsigned short*)d_ws;
    unsigned short* xbf = (unsigned short*)((char*)d_ws + X_OFF);

    hipMemsetAsync(d_ws, 0, WS_USED, stream);     // zero guards (~38 MB)
    prep_a<<<NCH, 256, 0, stream>>>(zero_, pos, gamma, abf);
    prep_x<<<H * BATCH * (SEQN / 32), 256, 0, stream>>>(x, xbf);
    toeplitz_wmma<<<dim3(NCH, 8), 128, 0, stream>>>(abf, xbf, out);
}
